// Processor_2628519985703
// MI455X (gfx1250) — matrix operand order repaired and checked by reading, not compiled, not run on hardware
//
#include <hip/hip_runtime.h>
#include <hip/hip_bf16.h>

// MI455X (gfx1250) implementation.
// Edge MLP: [E,384]x[384,128] -> relu -> [128,128] -> relu -> [128,128], residual,
//           scatter-add to agg (L2-resident, 25.6MB) via global_atomic_add_f32.
// Node MLP: [N,256]x[256,128] -> ... residual.
// GEMMs via v_wmma_f32_16x16x32_bf16 (wave32, one 16-row tile per wave).
// Weights are pre-swizzled into WMMA *fragment-major* order (t = k*8+n, lane-
// contiguous 32B chunks) so every B ds_load is base+lane*32 with a constant
// immediate offset t*1024. sched_barrier(0) pins the depth-2 B pipeline so the
// scheduler cannot sink loads past the consuming WMMA (keeps waits nonzero).

typedef __bf16 bf16_t;
typedef __attribute__((ext_vector_type(16))) __bf16 v16bf;
typedef __attribute__((ext_vector_type(8)))  __bf16 v8bf;
typedef __attribute__((ext_vector_type(4)))  __bf16 v4bf;
typedef __attribute__((ext_vector_type(8)))  float  v8f;

#define DFEAT 128
#define WPB   8      // waves per block (wave32)
#define TPB   256

// -------- weight prep: W[K][128] f32 -> fragment-major bf16 ------------------
// Fragment t = k*8 + n (k = K-tile of 32, n = 16-col tile). Within a fragment,
// lane L owns 16 bf16: element e -> B(K = k*32 + (e&8)*2 + (L>>4)*8 + (e&7), col = n*16 + (L&15)).
__global__ void wt_pack_kernel(const float* __restrict__ W,
                               bf16_t* __restrict__ out, int KDIM) {
  int i = blockIdx.x * blockDim.x + threadIdx.x;
  if (i < KDIM * DFEAT) {
    const int t  = i >> 9;          // /512 elems per fragment
    const int rr = i & 511;
    const int L  = rr >> 4;
    const int e  = rr & 15;
    const int k  = t >> 3;
    const int n  = t & 7;
    const int kk  = k * 32 + ((e < 8) ? (L >> 4) * 8 + e : 16 + (L >> 4) * 8 + (e - 8));
    const int col = n * 16 + (L & 15);
    out[i] = (bf16_t)W[(size_t)kk * DFEAT + col];
  }
}

// ---------------- one 16xKDIM @ KDIMx128 GEMM on a single wave ---------------
// X  : LDS, row-major [16][KDIM] bf16 (activation tile)
// Wf : LDS, fragment-major packed weights (see wt_pack_kernel)
// bs : per-lane bias values (bs[n] = bias[n*16 + (lane&15)]), preloaded.
template<int KDIM>
__device__ __forceinline__ void wave_gemm(const bf16_t* __restrict__ X,
                                          const bf16_t* __restrict__ Wf,
                                          const float bs[8],
                                          v8f acc[8], int lane) {
  const int row  = lane & 15;   // A row / D col index
  const int half = lane >> 4;
  constexpr int KT  = KDIM / 32;
  constexpr int TOT = 8 * KT;

  // init acc from bias (bias depends on column only -> splat per fragment)
#pragma unroll
  for (int n = 0; n < 8; ++n) {
    const float bb = bs[n];
#pragma unroll
    for (int i = 0; i < 8; ++i) acc[n][i] = bb;
  }

  // A fragment (16x32 bf16): elems 0..7 -> K = half*8+e ; 8..15 -> K = 16+half*8+e
  const bf16_t* xp  = X + row * KDIM + half * 8;
  // B fragments: lane-contiguous 32B chunks; constant offsets t*1024 (+16).
  const bf16_t* fb0 = Wf + lane * 16;
  // Second base for t >= 64 (only instantiated for KDIM=384). The offset is
  // laundered through an empty asm so the compiler keeps ONE base register and
  // folds (t-64)*1024 into the DS immediate instead of per-fragment adds.
  int hiOff = 64 * 512;
  if constexpr (TOT > 64) { asm("" : "+v"(hiOff)); }
  const bf16_t* fb1 = fb0 + hiOff;

  union Fu { v16bf v; v8bf h[2]; };
  Fu a[2];   // double-buffered A
  Fu b[3];   // rotating B (depth-2 pipeline)

#define LOAD_A(buf, kk)                                   \
  do {                                                    \
    (buf).h[0] = *(const v8bf*)(xp + (kk) * 32);          \
    (buf).h[1] = *(const v8bf*)(xp + (kk) * 32 + 16);     \
  } while (0)
#define LOAD_B(buf, tt)                                                        \
  do {                                                                         \
    const bf16_t* bp_ = ((tt) < 64) ? (fb0 + (tt) * 512)                       \
                                    : (fb1 + ((tt) - 64) * 512);               \
    (buf).h[0] = *(const v8bf*)(bp_);                                          \
    (buf).h[1] = *(const v8bf*)(bp_ + 8);                                      \
  } while (0)

  LOAD_A(a[0], 0);
  LOAD_B(b[0], 0);
  LOAD_B(b[1], 1);

#pragma unroll
  for (int t = 0; t < TOT; ++t) {
    const int n = t & 7;
    const int k = t >> 3;
    if (n == 0 && k + 1 < KT) LOAD_A(a[(k + 1) & 1], k + 1);   // overlap with 8 WMMAs
    if (t + 2 < TOT) LOAD_B(b[(t + 2) % 3], t + 2);            // B for t+2 in flight
    // Pin the schedule: loads above must issue BEFORE this WMMA; nothing may
    // cross. Keeps 2 fragments (4 ds_load_b128) in flight at every WMMA.
    __builtin_amdgcn_sched_barrier(0);
    acc[n] = __builtin_amdgcn_wmma_f32_16x16x32_bf16(false, a[k & 1].v,
                                                     false, b[t % 3].v,
                                                     (short)0, acc[n], false, false);
  }
#undef LOAD_A
#undef LOAD_B
}

// relu + f32->bf16, store D fragments to LDS tile [16][128] row-major
__device__ __forceinline__ void store_relu(const v8f acc[8], bf16_t* __restrict__ Y,
                                           int lane) {
  const int row  = lane & 15;
  const int half = lane >> 4;
#pragma unroll
  for (int n = 0; n < 8; ++n) {
#pragma unroll
    for (int i = 0; i < 8; ++i) {
      float v = acc[n][i];
      v = v > 0.f ? v : 0.f;
      Y[(i + half * 8) * DFEAT + n * 16 + row] = (bf16_t)v;
    }
  }
}

// ---------------------------- edge kernel -----------------------------------
__global__ __launch_bounds__(TPB, 1)
void edge_kernel(const float* __restrict__ x,
                 const float* __restrict__ edge_attr,
                 const int*   __restrict__ edge_index,
                 const bf16_t* __restrict__ wts,   // packed ew1|ew2|ew3 contiguous
                 const float* __restrict__ b1,
                 const float* __restrict__ b2,
                 const float* __restrict__ b3,
                 float* __restrict__ out_e,
                 float* __restrict__ agg,
                 int E, int nTiles) {
  extern __shared__ char smem[];
  bf16_t* w1 = (bf16_t*)smem;          // 128*384 (96 fragments)
  bf16_t* w2 = w1 + DFEAT * 3 * DFEAT; // 128*128 (32 fragments)
  bf16_t* w3 = w2 + DFEAT * DFEAT;
  char* wavesBase = (char*)(w3 + DFEAT * DFEAT);

  // stage all edge weights once per block: 163840 B = 10240 x int4
  {
    int4* dst = (int4*)smem;
    const int4* src = (const int4*)wts;
    for (int i = threadIdx.x; i < 10240; i += TPB) dst[i] = src[i];
  }
  __syncthreads();

  const int wave = threadIdx.x >> 5;
  const int lane = threadIdx.x & 31;
  const int row  = lane & 15;
  const int half = lane >> 4;

  // per-lane bias registers (invariant across tiles)
  float bs1[8], bs2[8], bs3[8];
#pragma unroll
  for (int n = 0; n < 8; ++n) {
    bs1[n] = b1[n * 16 + row];
    bs2[n] = b2[n * 16 + row];
    bs3[n] = b3[n * 16 + row];
  }

  char* pws = wavesBase + wave * (16 * 384 * 2 + 16 * 128 * 2 + 128);
  bf16_t* xcat = (bf16_t*)pws;                       // [16][384] concat tile
  bf16_t* hbuf = (bf16_t*)(pws + 16 * 384 * 2);      // [16][128] hidden (h1 then h2)
  int*    idxl = (int*)(pws + 16 * 384 * 2 + 16 * 128 * 2); // [32]: src[0..15], dst[0..15]

  const int tileStride = gridDim.x * WPB;
  for (int tile = blockIdx.x * WPB + wave; tile < nTiles; tile += tileStride) {
    const int base = tile * 16;
    idxl[lane] = (lane < 16) ? edge_index[base + lane]
                             : edge_index[E + base + (lane - 16)];

    // gather x[src] | x[dst] | edge_attr -> bf16 concat tile (float4 granularity)
    for (int it = lane; it < 16 * 96; it += 32) {
      int r  = it / 96;
      int c4 = it - r * 96;
      const float4* p;
      if (c4 < 32)
        p = (const float4*)(x + (size_t)idxl[r] * DFEAT) + c4;
      else if (c4 < 64)
        p = (const float4*)(x + (size_t)idxl[16 + r] * DFEAT) + (c4 - 32);
      else
        p = (const float4*)(edge_attr + (size_t)(base + r) * DFEAT) + (c4 - 64);
      float4 v = *p;
      v4bf o;
      o[0] = (bf16_t)v.x; o[1] = (bf16_t)v.y; o[2] = (bf16_t)v.z; o[3] = (bf16_t)v.w;
      *(v4bf*)(xcat + r * 384 + c4 * 4) = o;
    }

    // warm L2/WGP$ for the next tile's streaming operands (global_prefetch_b8)
    {
      const int nextTile = tile + tileStride;
      if (nextTile < nTiles) {
        if (lane < 16)
          __builtin_prefetch(edge_attr + (size_t)(nextTile * 16 + lane) * DFEAT, 0, 1);
        if (lane == 0)
          __builtin_prefetch(edge_index + nextTile * 16, 0, 1);
      }
    }

    v8f acc[8];
    wave_gemm<384>(xcat, w1, bs1, acc, lane);  // layer 1
    store_relu(acc, hbuf, lane);
    wave_gemm<128>(hbuf, w2, bs2, acc, lane);  // layer 2 (DS in-order: reads precede stores)
    store_relu(acc, hbuf, lane);
    wave_gemm<128>(hbuf, w3, bs3, acc, lane);  // layer 3 (no relu)

    // residual + store edge output + scatter-add to agg[dst]
#pragma unroll
    for (int n = 0; n < 8; ++n) {
      const int col = n * 16 + row;
#pragma unroll
      for (int i = 0; i < 8; ++i) {
        const int m = i + half * 8;
        const size_t e = (size_t)(base + m);
        float v = acc[n][i] + edge_attr[e * DFEAT + col];   // L2 hit (same tile)
        out_e[e * DFEAT + col] = v;
        const int dn = idxl[16 + m];
        __hip_atomic_fetch_add(&agg[(size_t)dn * DFEAT + col], v,
                               __ATOMIC_RELAXED, __HIP_MEMORY_SCOPE_AGENT);
      }
    }
  }
}

// ---------------------------- node kernel -----------------------------------
__global__ __launch_bounds__(TPB, 1)
void node_kernel(const float* __restrict__ x,
                 const float* __restrict__ agg,
                 const bf16_t* __restrict__ wts,   // packed nw1|nw2|nw3 contiguous
                 const float* __restrict__ b1,
                 const float* __restrict__ b2,
                 const float* __restrict__ b3,
                 float* __restrict__ out_x,
                 int nTiles) {
  extern __shared__ char smem[];
  bf16_t* w1 = (bf16_t*)smem;          // 128*256 (64 fragments)
  bf16_t* w2 = w1 + DFEAT * 2 * DFEAT;
  bf16_t* w3 = w2 + DFEAT * DFEAT;
  char* wavesBase = (char*)(w3 + DFEAT * DFEAT);

  // stage node weights: 131072 B = 8192 x int4
  {
    int4* dst = (int4*)smem;
    const int4* src = (const int4*)wts;
    for (int i = threadIdx.x; i < 8192; i += TPB) dst[i] = src[i];
  }
  __syncthreads();

  const int wave = threadIdx.x >> 5;
  const int lane = threadIdx.x & 31;
  const int row  = lane & 15;
  const int half = lane >> 4;

  float bs1[8], bs2[8], bs3[8];
#pragma unroll
  for (int n = 0; n < 8; ++n) {
    bs1[n] = b1[n * 16 + row];
    bs2[n] = b2[n * 16 + row];
    bs3[n] = b3[n * 16 + row];
  }

  char* pws = wavesBase + wave * (16 * 256 * 2 + 16 * 128 * 2);
  bf16_t* xcat = (bf16_t*)pws;                  // [16][256]
  bf16_t* hbuf = (bf16_t*)(pws + 16 * 256 * 2); // [16][128]

  const int tileStride = gridDim.x * WPB;
  for (int tile = blockIdx.x * WPB + wave; tile < nTiles; tile += tileStride) {
    const int base = tile * 16;

    // concat [x, agg] -> bf16 tile
    for (int it = lane; it < 16 * 64; it += 32) {
      int r  = it / 64;
      int c4 = it - r * 64;
      const size_t node = (size_t)(base + r);
      const float4* p = (c4 < 32)
          ? (const float4*)(x + node * DFEAT) + c4
          : (const float4*)(agg + node * DFEAT) + (c4 - 32);
      float4 v = *p;
      v4bf o;
      o[0] = (bf16_t)v.x; o[1] = (bf16_t)v.y; o[2] = (bf16_t)v.z; o[3] = (bf16_t)v.w;
      *(v4bf*)(xcat + r * 256 + c4 * 4) = o;
    }

    {
      const int nextTile = tile + tileStride;
      if (nextTile < nTiles) {
        const size_t nn = (size_t)(nextTile * 16) + (lane & 15);
        __builtin_prefetch((lane < 16 ? x : agg) + nn * DFEAT, 0, 1);
      }
    }

    v8f acc[8];
    wave_gemm<256>(xcat, w1, bs1, acc, lane);
    store_relu(acc, hbuf, lane);
    wave_gemm<128>(hbuf, w2, bs2, acc, lane);
    store_relu(acc, hbuf, lane);
    wave_gemm<128>(hbuf, w3, bs3, acc, lane);

#pragma unroll
    for (int n = 0; n < 8; ++n) {
      const int col = n * 16 + row;
#pragma unroll
      for (int i = 0; i < 8; ++i) {
        const int m = i + half * 8;
        const size_t node = (size_t)(base + m);
        out_x[node * DFEAT + col] = acc[n][i] + x[node * DFEAT + col];
      }
    }
  }
}

// ---------------------------- launcher --------------------------------------
extern "C" void kernel_launch(void* const* d_in, const int* in_sizes, int n_in,
                              void* d_out, int out_size, void* d_ws, size_t ws_size,
                              hipStream_t stream) {
  const float* x    = (const float*)d_in[0];
  const float* ea   = (const float*)d_in[1];
  const int*   eidx = (const int*)d_in[2];
  const float* ew1  = (const float*)d_in[3];
  const float* eb1  = (const float*)d_in[4];
  const float* ew2  = (const float*)d_in[5];
  const float* eb2  = (const float*)d_in[6];
  const float* ew3  = (const float*)d_in[7];
  const float* eb3  = (const float*)d_in[8];
  const float* nw1  = (const float*)d_in[9];
  const float* nb1  = (const float*)d_in[10];
  const float* nw2  = (const float*)d_in[11];
  const float* nb2  = (const float*)d_in[12];
  const float* nw3  = (const float*)d_in[13];
  const float* nb3  = (const float*)d_in[14];

  const int N = in_sizes[0] / DFEAT;   // 50000
  const int E = in_sizes[1] / DFEAT;   // 600000

  float* out_x = (float*)d_out;
  float* out_e = out_x + (size_t)N * DFEAT;

  char* ws = (char*)d_ws;
  float* agg = (float*)ws;                         // N*128 f32 (L2-resident)
  const size_t aggBytes = (size_t)N * DFEAT * sizeof(float);
  bf16_t* ewt1 = (bf16_t*)(ws + aggBytes);         // fragment-packed bf16 weights
  bf16_t* ewt2 = ewt1 + DFEAT * 3 * DFEAT;
  bf16_t* ewt3 = ewt2 + DFEAT * DFEAT;
  bf16_t* nwt1 = ewt3 + DFEAT * DFEAT;
  bf16_t* nwt2 = nwt1 + DFEAT * 2 * DFEAT;
  bf16_t* nwt3 = nwt2 + DFEAT * DFEAT;

  hipMemsetAsync(agg, 0, aggBytes, stream);

  struct { const float* W; bf16_t* Wt; int K; } cv[6] = {
    {ew1, ewt1, 3 * DFEAT}, {ew2, ewt2, DFEAT}, {ew3, ewt3, DFEAT},
    {nw1, nwt1, 2 * DFEAT}, {nw2, nwt2, DFEAT}, {nw3, nwt3, DFEAT},
  };
  for (int i = 0; i < 6; ++i) {
    int total = cv[i].K * DFEAT;
    wt_pack_kernel<<<(total + 255) / 256, 256, 0, stream>>>(cv[i].W, cv[i].Wt, cv[i].K);
  }

  const int eTiles = E / 16;   // 37500
  const int nTiles = N / 16;   // 3125
  const int eGrid = (eTiles + WPB * 4 - 1) / (WPB * 4);  // ~4 tiles per wave
  const int nGrid = (nTiles + WPB * 4 - 1) / (WPB * 4);
  const size_t eSm = 163840 + (size_t)WPB * (16 * 384 * 2 + 16 * 128 * 2 + 128); // 295936 B
  const size_t nSm = 131072 + (size_t)WPB * (16 * 256 * 2 + 16 * 128 * 2);       // 229376 B

  edge_kernel<<<eGrid, TPB, eSm, stream>>>(x, ea, eidx, ewt1, eb1, eb2, eb3,
                                           out_e, agg, E, eTiles);
  node_kernel<<<nGrid, TPB, nSm, stream>>>(x, agg, nwt1, nb1, nb2, nb3,
                                           out_x, nTiles);
}
